// DynamicCacheAttention_39779987096355
// MI455X (gfx1250) — matrix operand order repaired and checked by reading, not compiled, not run on hardware
//
#include <hip/hip_runtime.h>
#include <hip/hip_fp16.h>

#define HID 4096
#define B_  4
#define T_  16
#define H_  32
#define D_  128
#define SPAST 4096
#define STOT (SPAST + T_)               // 4112
#define NSEG 4
#define SEG_LEN ((STOT + NSEG - 1) / NSEG)  // 1028
#define CHUNK 128

typedef __attribute__((ext_vector_type(16))) _Float16 v16h;
typedef __attribute__((ext_vector_type(8)))  float    v8f;

union FragAB { v16h v; _Float16 h[16]; };
union FragC  { v8f  v; float    f[8];  };

// ---------------------------------------------------------------------------
// Kernel 0: convert hidden f32 -> f16 (tiny: 1 MB read)
// ---------------------------------------------------------------------------
__global__ __launch_bounds__(256) void cvt_hidden_kernel(
    const float* __restrict__ x, _Float16* __restrict__ y, int n) {
  int idx = (blockIdx.x * blockDim.x + threadIdx.x) * 4;
  if (idx + 3 < n) {
    float4 f = *(const float4*)(x + idx);
    y[idx + 0] = (_Float16)f.x;
    y[idx + 1] = (_Float16)f.y;
    y[idx + 2] = (_Float16)f.z;
    y[idx + 3] = (_Float16)f.w;
  }
}

// ---------------------------------------------------------------------------
// Kernel 1: Y[64, HID] = A[64, HID](f16) @ W[HID, HID]^T (W rows are out cols)
// 8 waves; each wave owns one 16-row M tile and 4 N tiles of 16, K step 32.
// mode 0: f32 row-major [64][HID]; mode 1: f32 k/v_full tail [B,H,STOT,D];
// mode 2: f16 qbuf [B,H,T,D]
// ---------------------------------------------------------------------------
__global__ __launch_bounds__(256) void gemm64_kernel(
    const _Float16* __restrict__ A, const float* __restrict__ W,
    float* __restrict__ outF32, _Float16* __restrict__ outF16, int mode) {
  const int lane  = threadIdx.x & 31;
  const int wave  = threadIdx.x >> 5;
  const int lm    = lane & 15;
  const int khalf = (lane < 16) ? 0 : 8;   // unused; the A frag is loaded with kb
  const int kb    = (lane < 16) ? 0 : 16;  // A/B-frag K offset (contiguous k = 16h + e)

  const int n0    = blockIdx.x * 128;
  const int m0    = (wave >> 1) * 16;
  const int nbase = n0 + (wave & 1) * 64;

  FragC acc[4];
  for (int j = 0; j < 4; ++j)
    for (int e = 0; e < 8; ++e) acc[j].f[e] = 0.0f;

  for (int k0 = 0; k0 < HID; k0 += 32) {
    FragAB a;
    const _Float16* arow = A + (size_t)(m0 + lm) * HID + k0;
    for (int e = 0; e < 16; ++e) a.h[e] = arow[kb + e];
    for (int j = 0; j < 4; ++j) {
      const int n = nbase + j * 16 + lm;
      const float* wrow = W + (size_t)n * HID + k0 + kb;
      FragAB b;
      for (int e = 0; e < 16; ++e) b.h[e] = (_Float16)wrow[e];
      acc[j].v = __builtin_amdgcn_wmma_f32_16x16x32_f16(
          false, a.v, false, b.v, (short)0, acc[j].v, false, false);
    }
  }

  for (int j = 0; j < 4; ++j) {
    const int ncol = nbase + j * 16 + lm;
    for (int r = 0; r < 8; ++r) {
      const int m = m0 + r + ((lane < 16) ? 0 : 8);  // C layout: VGPR r -> M
      const float val = acc[j].f[r];
      if (mode == 0) {
        outF32[(size_t)m * HID + ncol] = val;
      } else {
        const int b = m >> 4, t = m & 15;
        const int h = ncol >> 7, d = ncol & 127;
        if (mode == 1)
          outF32[(((size_t)(b * H_ + h) * STOT) + (SPAST + t)) * D_ + d] = val;
        else
          outF16[(((size_t)(b * H_ + h) * T_) + t) * D_ + d] = (_Float16)val;
      }
    }
  }
}

// ---------------------------------------------------------------------------
// Kernel 2: fused cache-copy + flash attention (segmented, async data path).
// grid = NSEG*128, block = 256 (8 waves). Per 128-row K/V chunk:
//   global f32 --(GLOBAL_LOAD_ASYNC_TO_LDS_B128)--> LDS f32 tiles
//   LDS f32   --(GLOBAL_STORE_ASYNC_FROM_LDS_B128)--> k_full/v_full (concat)
//   LDS f32   --> f16 fragments (K row-major; V via transposed f16 tile)
//   WMMA scores -> cross-wave online softmax -> WMMA PV
// ASYNCcnt drained with s_wait_asynccnt; barriers publish tiles across waves.
// ---------------------------------------------------------------------------
__global__ __launch_bounds__(256) void attn_kernel(
    const float* __restrict__ cache_k, const float* __restrict__ cache_v,
    const _Float16* __restrict__ qbuf, float* __restrict__ k_full,
    float* __restrict__ v_full, float* __restrict__ m_part,
    float* __restrict__ l_part, float* __restrict__ ctx_part) {
  const int bh   = blockIdx.x & 127;
  const int seg  = blockIdx.x >> 7;
  const int tid  = threadIdx.x;
  const int lane = tid & 31;
  const int wave = tid >> 5;
  const int lm   = lane & 15;
  const int khalf = (lane < 16) ? 0 : 8;
  const int kb    = (lane < 16) ? 0 : 16;

  __shared__ float    KtF32[CHUNK][132];   // f32 K tile (async in/out)
  __shared__ float    VtF32[CHUNK][132];   // f32 V tile (async in/out)
  __shared__ _Float16 VtT[D_][136];        // f16 V transposed [d][s]
  __shared__ _Float16 Qt[16][136];
  __shared__ _Float16 Pt[16][136];
  __shared__ float    St[16][128];
  __shared__ float    red[16][16];
  __shared__ float    mrow[16], lrow[16], arow[16];

  {  // load Q tile (f16) once
    const _Float16* q = qbuf + (size_t)bh * (T_ * D_);
    for (int i = tid; i < 16 * 128; i += 256) Qt[i >> 7][i & 127] = q[i];
    if (tid < 16) { mrow[tid] = -3.0e38f; lrow[tid] = 0.0f; }
  }

  FragC ctx;
  for (int e = 0; e < 8; ++e) ctx.f[e] = 0.0f;

  const int seg_start = seg * SEG_LEN;
  const int seg_end   = (seg_start + SEG_LEN < STOT) ? (seg_start + SEG_LEN) : STOT;
  const float scale = 0.08838834764831845f;  // 1/sqrt(128)

  const float* ck = cache_k + (size_t)bh * SPAST * D_;
  const float* cv = cache_v + (size_t)bh * SPAST * D_;
  float* kf = k_full + (size_t)bh * STOT * D_;
  float* vf = v_full + (size_t)bh * STOT * D_;

  for (int cs = seg_start; cs < seg_end; cs += CHUNK) {
    // Drain my previous copy-out stores, then join so no wave's store still
    // reads the f32 tiles we are about to overwrite.
    asm volatile("s_wait_asynccnt 0x0" ::: "memory");
    __syncthreads();

    {  // async load K/V chunk: 4096 x b128 transfers / 256 threads = 16 iters
      for (int it = 0; it < 16; ++it) {
        const int cid = it * 256 + tid;       // 0..4095
        const int r   = cid >> 5;             // tile row 0..127
        const int c4  = (cid & 31) * 4;       // float col 0,4,...,124
        const int s   = cs + r;
        const unsigned ldsK = (unsigned)(size_t)&KtF32[r][c4];
        const unsigned ldsV = (unsigned)(size_t)&VtF32[r][c4];
        if (s < STOT) {
          // past rows stream from the cache; the 16 new rows were written into
          // k_full/v_full by the projection GEMM already.
          const float* srcK = (s < SPAST) ? (ck + (size_t)s * D_ + c4)
                                          : (kf + (size_t)s * D_ + c4);
          const float* srcV = (s < SPAST) ? (cv + (size_t)s * D_ + c4)
                                          : (vf + (size_t)s * D_ + c4);
          const unsigned long long gk = (unsigned long long)(size_t)srcK;
          const unsigned long long gv = (unsigned long long)(size_t)srcV;
          asm volatile("global_load_async_to_lds_b128 %0, %1, off"
                       :: "v"(ldsK), "v"(gk) : "memory");
          asm volatile("global_load_async_to_lds_b128 %0, %1, off"
                       :: "v"(ldsV), "v"(gv) : "memory");
        } else {  // keep WMMA inputs finite for masked columns
          float4 z = make_float4(0.f, 0.f, 0.f, 0.f);
          *(float4*)&KtF32[r][c4] = z;
          *(float4*)&VtF32[r][c4] = z;
        }
      }
      asm volatile("s_wait_asynccnt 0x0" ::: "memory");
    }
    __syncthreads();

    {  // fused DynamicCache concat: async store LDS f32 tiles -> k_full/v_full
       // (read-only on the tiles; overlaps with the compute below)
      for (int it = 0; it < 16; ++it) {
        const int cid = it * 256 + tid;
        const int r   = cid >> 5;
        const int c4  = (cid & 31) * 4;
        const int s   = cs + r;
        if (s < SPAST) {
          const unsigned ldsK = (unsigned)(size_t)&KtF32[r][c4];
          const unsigned ldsV = (unsigned)(size_t)&VtF32[r][c4];
          const unsigned long long gk =
              (unsigned long long)(size_t)(kf + (size_t)s * D_ + c4);
          const unsigned long long gv =
              (unsigned long long)(size_t)(vf + (size_t)s * D_ + c4);
          asm volatile("global_store_async_from_lds_b128 %0, %1, off"
                       :: "v"(gk), "v"(ldsK) : "memory");
          asm volatile("global_store_async_from_lds_b128 %0, %1, off"
                       :: "v"(gv), "v"(ldsV) : "memory");
        }
      }
    }

    {  // build transposed f16 V tile: VtT[d][s] (each thread reads 64 contiguous f32 of one row)
      for (int i = 0; i < 64; ++i) {
        const int e = tid * 64 + i;           // 0..16383
        const int s = e >> 7, d = e & 127;
        VtT[d][s] = (_Float16)VtF32[s][d];
      }
    }

    // scores: wave w owns score columns [w*16, w*16+16); K-dim = d (4 x 32)
    FragC sc;
    for (int e = 0; e < 8; ++e) sc.f[e] = 0.0f;
    const int srow = wave * 16 + lm;
    for (int d0 = 0; d0 < D_; d0 += 32) {
      FragAB a, b;
      for (int e = 0; e < 16; ++e) a.h[e]    = Qt[lm][d0 + kb + e];
      for (int e = 0; e < 16; ++e) b.h[e]    = (_Float16)KtF32[srow][d0 + kb + e];
      sc.v = __builtin_amdgcn_wmma_f32_16x16x32_f16(
          false, a.v, false, b.v, (short)0, sc.v, false, false);
    }
    {  // stage scaled+masked scores for cross-wave softmax
      const int col = wave * 16 + lm;
      const bool ok = (cs + col) < seg_end;
      for (int r = 0; r < 8; ++r) {
        const int m = r + ((lane < 16) ? 0 : 8);
        St[m][col] = ok ? sc.f[r] * scale : -3.0e38f;
      }
    }
    __syncthreads();  // publishes St and VtT

    {  // row max: 16 threads per row, 8 cols each
      const int row = tid >> 4, sub = tid & 15;
      float mx = -3.0e38f;
      for (int i = 0; i < 8; ++i) mx = fmaxf(mx, St[row][sub * 8 + i]);
      red[row][sub] = mx;
    }
    __syncthreads();
    if (tid < 16) {
      float mc = red[tid][0];
      for (int i = 1; i < 16; ++i) mc = fmaxf(mc, red[tid][i]);
      const float mo = mrow[tid];
      const float mn = fmaxf(mo, mc);
      arow[tid] = __expf(mo - mn);
      mrow[tid] = mn;
    }
    __syncthreads();
    {  // exponentiate, write P (f16), partial row sums
      const int row = tid >> 4, sub = tid & 15;
      const float mn = mrow[row];
      float ls = 0.0f;
      for (int i = 0; i < 8; ++i) {
        const int c = sub * 8 + i;
        const float p = __expf(St[row][c] - mn);
        Pt[row][c] = (_Float16)p;
        ls += p;
      }
      red[row][sub] = ls;
    }
    __syncthreads();
    if (tid < 16) {
      float s = 0.0f;
      for (int i = 0; i < 16; ++i) s += red[tid][i];
      lrow[tid] = lrow[tid] * arow[tid] + s;
    }
    __syncthreads();

    // rescale running ctx, then ctx += P @ V (wave w owns d cols [w*16, +16))
    for (int r = 0; r < 8; ++r) {
      const int m = r + ((lane < 16) ? 0 : 8);
      ctx.f[r] *= arow[m];
    }
    const int dcol = wave * 16 + lm;
    for (int s0 = 0; s0 < CHUNK; s0 += 32) {
      FragAB a, b;
      for (int e = 0; e < 16; ++e) a.h[e]    = Pt[lm][s0 + kb + e];
      for (int e = 0; e < 16; ++e) b.h[e]    = VtT[dcol][s0 + kb + e];
      ctx.v = __builtin_amdgcn_wmma_f32_16x16x32_f16(
          false, a.v, false, b.v, (short)0, ctx.v, false, false);
    }
  }

  __syncthreads();
  {  // emit segment partials (trailing async stores drained by s_endpgm)
    float* cp = ctx_part + ((size_t)seg * 128 + bh) * 16 * 128;
    const int dcol = wave * 16 + lm;
    for (int r = 0; r < 8; ++r) {
      const int m = r + ((lane < 16) ? 0 : 8);
      cp[m * 128 + dcol] = ctx.f[r];
    }
    if (tid < 16) {
      m_part[(seg * 128 + bh) * 16 + tid] = mrow[tid];
      l_part[(seg * 128 + bh) * 16 + tid] = lrow[tid];
    }
  }
}

// ---------------------------------------------------------------------------
// Kernel 3: combine NSEG partials -> normalized ctx, layout [64][HID] f16
// ---------------------------------------------------------------------------
__global__ __launch_bounds__(128) void combine_kernel(
    const float* __restrict__ m_part, const float* __restrict__ l_part,
    const float* __restrict__ ctx_part, _Float16* __restrict__ ctxbuf) {
  const int bh = blockIdx.x, d = threadIdx.x;
  const int b = bh >> 5, h = bh & 31;
  for (int t = 0; t < 16; ++t) {
    float ms = -3.0e38f;
    for (int s = 0; s < NSEG; ++s)
      ms = fmaxf(ms, m_part[(s * 128 + bh) * 16 + t]);
    float lsum = 0.0f, acc = 0.0f;
    for (int s = 0; s < NSEG; ++s) {
      const float w = __expf(m_part[(s * 128 + bh) * 16 + t] - ms);
      lsum += w * l_part[(s * 128 + bh) * 16 + t];
      acc  += w * ctx_part[(((size_t)s * 128 + bh) * 16 + t) * 128 + d];
    }
    const int m = b * 16 + t;
    ctxbuf[(size_t)m * HID + h * 128 + d] = (_Float16)(acc / lsum);
  }
}

// ---------------------------------------------------------------------------
extern "C" void kernel_launch(void* const* d_in, const int* in_sizes, int n_in,
                              void* d_out, int out_size, void* d_ws,
                              size_t ws_size, hipStream_t stream) {
  const float* hidden  = (const float*)d_in[0];
  const float* cache_k = (const float*)d_in[1];
  const float* cache_v = (const float*)d_in[2];
  const float* wq = (const float*)d_in[3];
  const float* wk = (const float*)d_in[4];
  const float* wv = (const float*)d_in[5];
  const float* wo = (const float*)d_in[6];

  float* out    = (float*)d_out;                       // [B,T,HID]
  float* k_full = out + (size_t)B_ * T_ * HID;         // [B,H,STOT,D]
  float* v_full = k_full + (size_t)B_ * H_ * STOT * D_;

  char* ws = (char*)d_ws;
  _Float16* hid16   = (_Float16*)(ws);                  // 512 KB
  _Float16* qbuf    = (_Float16*)(ws + 524288);         // 512 KB
  _Float16* ctxbuf  = (_Float16*)(ws + 2 * 524288);     // 512 KB
  float*    m_part  = (float*)(ws + 3 * 524288);        // 32 KB
  float*    l_part  = (float*)(ws + 3 * 524288 + 32768);
  float*    ctx_prt = (float*)(ws + 3 * 524288 + 65536);// 4 MB

  cvt_hidden_kernel<<<256, 256, 0, stream>>>(hidden, hid16, 64 * HID);
  gemm64_kernel<<<32, 256, 0, stream>>>(hid16, wq, nullptr, qbuf, 2);
  gemm64_kernel<<<32, 256, 0, stream>>>(hid16, wk, k_full, nullptr, 1);
  gemm64_kernel<<<32, 256, 0, stream>>>(hid16, wv, v_full, nullptr, 1);
  attn_kernel<<<NSEG * 128, 256, 0, stream>>>(cache_k, cache_v, qbuf, k_full,
                                              v_full, m_part, l_part, ctx_prt);
  combine_kernel<<<128, 128, 0, stream>>>(m_part, l_part, ctx_prt, ctxbuf);
  gemm64_kernel<<<32, 256, 0, stream>>>(ctxbuf, wo, out, nullptr, 0);
}
